// RecurrentHierarchicalTransformer_25804163514522
// MI455X (gfx1250) — matrix operand order repaired and checked by reading, not compiled, not run on hardware
//
#include <hip/hip_runtime.h>
#include <hip/hip_bf16.h>
#include <math.h>

// ---- problem constants (from reference) ----
#define B_    4
#define C_    32
#define T_    1000
#define PL_   20
#define NW_   50
#define D_    128
#define H_    8
#define DH_   64
#define HD_   512
#define L_    4
#define F_    512
#define NCLS_ 2
#define N_    1601   // C*NW + 1
#define NPAD_ 1632   // N rounded up to 32 (zero-padded K for attn@V B-operand)

typedef __attribute__((ext_vector_type(16))) __bf16 v16bf;
typedef __attribute__((ext_vector_type(8)))  __bf16 bf16x8;
typedef __attribute__((ext_vector_type(8)))  float  v8f;

// epilogue modes
#define EPI_STORE     0
#define EPI_BIAS      1
#define EPI_BIAS_RES  2
#define EPI_GELU_BIAS 3
#define EPI_SHIFT     4

__device__ __forceinline__ v16bf pack16(bf16x8 lo, bf16x8 hi) {
  return __builtin_shufflevector(lo, hi, 0,1,2,3,4,5,6,7,8,9,10,11,12,13,14,15);
}

// ---------------------------------------------------------------------------
// Attention-score GEMM (M = N = Nseq, K = 64), batched over (b,h).
// Block computes a 64(M) x 128(N) C-tile: A panel 64x64 and B panel 128x64 are
// staged in LDS with GLOBAL_LOAD_ASYNC_TO_LDS_B128 (ASYNCcnt path), then each
// of the 8 waves computes a 32x32 sub-tile (2x2 WMMA accumulators) from LDS.
// L2 traffic per block: 24KB for 1.05 MFLOP -> ~44 flop/byte (vs ~8 direct).
// ---------------------------------------------------------------------------
template<int EPI>
__global__ __launch_bounds__(256)
void score_gemm(const __bf16* __restrict__ Q,  int ldq, long qOffB, long qOffH,
                const __bf16* __restrict__ Kt, int ldk, long kOffB, long kOffH,
                float* __restrict__ Cl, int Nseq)
{
  const int z  = blockIdx.z;
  const int zb = z >> 3, zh = z & 7;          // H_ == 8
  Q  += (long)zb * qOffB + (long)zh * qOffH;
  Kt += (long)zb * kOffB + (long)zh * kOffH;
  Cl += (long)z * Nseq * Nseq;

  const int t    = threadIdx.x;
  const int wave = t >> 5, lane = t & 31;
  const int m0 = blockIdx.y * 64;
  const int n0 = blockIdx.x * 128;

  __shared__ __align__(16) __bf16 sA[64 * 64];    //  8 KB, A[r][k]
  __shared__ __align__(16) __bf16 sB[128 * 64];   // 16 KB, B[c][k]

  // ---- async stage A panel: thread t covers row t/4, k = (t%4)*16 .. +15 ----
  {
    int r  = t >> 2;
    int kk = (t & 3) * 16;
    int gr = m0 + r; gr = gr < Nseq ? gr : Nseq - 1;   // clamp: junk rows unused
    const __bf16* g = Q + (long)gr * ldq + kk;
    unsigned l = (unsigned)(uintptr_t)&sA[r * 64 + kk]; // low 32 bits == DS addr
    asm volatile("global_load_async_to_lds_b128 %0, %1, off"
                 :: "v"(l), "v"(g) : "memory");
    asm volatile("global_load_async_to_lds_b128 %0, %1, off offset:16"
                 :: "v"(l), "v"(g) : "memory");        // INST_OFFSET hits both sides
  }
  // ---- async stage B panel: thread t covers col t/2, k = (t%2)*32 .. +31 ----
  {
    int c  = t >> 1;
    int kk = (t & 1) * 32;
    int gc = n0 + c; gc = gc < Nseq ? gc : Nseq - 1;
    const __bf16* g = Kt + (long)gc * ldk + kk;
    unsigned l = (unsigned)(uintptr_t)&sB[c * 64 + kk];
    asm volatile("global_load_async_to_lds_b128 %0, %1, off"
                 :: "v"(l), "v"(g) : "memory");
    asm volatile("global_load_async_to_lds_b128 %0, %1, off offset:16"
                 :: "v"(l), "v"(g) : "memory");
    asm volatile("global_load_async_to_lds_b128 %0, %1, off offset:32"
                 :: "v"(l), "v"(g) : "memory");
    asm volatile("global_load_async_to_lds_b128 %0, %1, off offset:48"
                 :: "v"(l), "v"(g) : "memory");
  }
  asm volatile("s_wait_asynccnt 0" ::: "memory");
  __syncthreads();

  // ---- compute: wave (wm,wn) does rows wm*32..+31, cols wn*32..+31 ----
  const int wm = wave >> 2, wn = wave & 3;
  const int akb  = (lane >> 4) * 8;
  const int bkb  = (lane >> 4) * 8;
  const int lrow = lane & 15;

  const v8f zero = {0.f, 0.f, 0.f, 0.f, 0.f, 0.f, 0.f, 0.f};
  v8f acc[2][2] = {{zero, zero}, {zero, zero}};

#pragma unroll
  for (int k0 = 0; k0 < 64; k0 += 32) {
    v16bf a[2], b[2];
#pragma unroll
    for (int ms = 0; ms < 2; ++ms) {
      const __bf16* p = &sA[(wm * 32 + ms * 16 + lrow) * 64 + akb + k0];
      a[ms] = pack16(*(const bf16x8*)p, *(const bf16x8*)(p + 16));
    }
#pragma unroll
    for (int ns = 0; ns < 2; ++ns) {
      const __bf16* p = &sB[(wn * 32 + ns * 16 + lrow) * 64 + bkb + k0];
      b[ns] = pack16(*(const bf16x8*)p, *(const bf16x8*)(p + 16));
    }
#pragma unroll
    for (int ms = 0; ms < 2; ++ms)
#pragma unroll
      for (int ns = 0; ns < 2; ++ns)
        acc[ms][ns] = __builtin_amdgcn_wmma_f32_16x16x32_bf16(
            false, a[ms], false, b[ns], (short)0, acc[ms][ns], false, false);
  }

  // ---- epilogue ----
#pragma unroll
  for (int ms = 0; ms < 2; ++ms) {
#pragma unroll
    for (int ns = 0; ns < 2; ++ns) {
      int ccol  = n0 + wn * 32 + ns * 16 + lrow;
      int crow0 = m0 + wm * 32 + ms * 16 + ((lane >> 4) ? 8 : 0);
#pragma unroll
      for (int g = 0; g < 8; ++g) {
        int r = crow0 + g;
        if (r >= Nseq || ccol >= Nseq) continue;
        float v = acc[ms][ns][g];
        if constexpr (EPI == EPI_STORE) {
          Cl[(long)r * Nseq + ccol] = v;
        } else {  // EPI_SHIFT: rel_shift scatter (bijective, race-free +=)
          float hv = 0.5f * v;
          int c1 = ccol + r + 1;
          if (r >= 1 && c1 < Nseq) Cl[(long)(r - 1) * Nseq + c1] += hv;
          int off = Nseq - 1 - r;
          if (ccol >= off) Cl[(long)r * Nseq + (ccol - off)] += hv;
        }
      }
    }
  }
}

// ---------------------------------------------------------------------------
// Generic bf16 WMMA GEMM ("NT": B stored Ncols x K). One wave per 16x16 tile.
// Out-of-range rows/cols clamped (junk rows/cols never stored) -> no in-loop
// predication. K tail (attn@V, K=1601): A value-selected, B zero-padded rows.
// ---------------------------------------------------------------------------
template<typename AT, typename OT, int EPI>
__global__ __launch_bounds__(256)
void wmma_gemm(const AT* __restrict__ A,      int lda, long aOffB, long aOffH,
               const __bf16* __restrict__ Bm, int ldb, long bOffB, long bOffH,
               OT* __restrict__ C,            int ldc, long cOffB, long cOffH,
               int M, int Ncols, int K, int Hdiv,
               const float* __restrict__ bias, int Nseq)
{
  const int z  = blockIdx.z;
  const int zb = z / Hdiv, zh = z % Hdiv;
  A  += (long)zb * aOffB + (long)zh * aOffH;
  Bm += (long)zb * bOffB + (long)zh * bOffH;
  C  += (long)zb * cOffB + (long)zh * cOffH;

  const int wave = threadIdx.x >> 5;
  const int lane = threadIdx.x & 31;
  const int m0 = blockIdx.y * 32 + (wave >> 2) * 16;
  const int n0 = blockIdx.x * 64 + (wave & 3) * 16;
  if (m0 >= M || n0 >= Ncols) return;   // wave-uniform: EXEC stays all-ones

  const int arow = m0 + (lane & 15);
  const int akb  = (lane >> 4) * 8;
  const int bcol = n0 + (lane & 15);
  const int bkb  = (lane >> 4) * 8;
  const int arowc = arow < M ? arow : M - 1;
  const int bcolc = bcol < Ncols ? bcol : Ncols - 1;

  const AT*     pa = A  + (long)arowc * lda + akb;
  const __bf16* pb = Bm + (long)bcolc * ldb + bkb;

  v8f acc = {0.f, 0.f, 0.f, 0.f, 0.f, 0.f, 0.f, 0.f};
  const int K32 = K & ~31;

  for (int k0 = 0; k0 < K32; k0 += 32) {
    v16bf av;
    if constexpr (__is_same(AT, __bf16)) {
      av = pack16(*(const bf16x8*)(pa + k0), *(const bf16x8*)(pa + k0 + 16));
    } else {
#pragma unroll
      for (int e = 0; e < 8; ++e) {
        av[e]     = (__bf16)pa[k0 + e];
        av[e + 8] = (__bf16)pa[k0 + 16 + e];
      }
    }
    v16bf bv = pack16(*(const bf16x8*)(pb + k0), *(const bf16x8*)(pb + k0 + 16));
    acc = __builtin_amdgcn_wmma_f32_16x16x32_bf16(
        false, av, false, bv, (short)0, acc, false, false);
  }

  if (K32 < K) {  // single tail iteration (attn@V only)
    v16bf av;
#pragma unroll
    for (int e = 0; e < 16; ++e) {
      int off = K32 + e + (e >= 8 ? 8 : 0);
      int ka  = akb + off;
      AT raw  = pa[ka < K ? off : 0];
      av[e]   = (ka < K) ? (__bf16)(float)raw : (__bf16)0.f;
    }
    v16bf bv = pack16(*(const bf16x8*)(pb + K32), *(const bf16x8*)(pb + K32 + 16));
    acc = __builtin_amdgcn_wmma_f32_16x16x32_bf16(
        false, av, false, bv, (short)0, acc, false, false);
  }

  const int ccol  = n0 + (lane & 15);
  const int crow0 = m0 + ((lane >> 4) ? 8 : 0);
#pragma unroll
  for (int g = 0; g < 8; ++g) {
    int r = crow0 + g;
    if (r >= M || ccol >= Ncols) continue;
    float v = acc[g];
    if constexpr (EPI == EPI_STORE) {
      C[(long)r * ldc + ccol] = (OT)v;
    } else if constexpr (EPI == EPI_BIAS) {
      C[(long)r * ldc + ccol] = (OT)(v + bias[ccol]);
    } else if constexpr (EPI == EPI_BIAS_RES) {
      C[(long)r * ldc + ccol] += v + bias[ccol];
    } else if constexpr (EPI == EPI_GELU_BIAS) {
      float tt = v + bias[ccol];
      C[(long)r * ldc + ccol] = (OT)(0.5f * tt * (1.f + erff(tt * 0.70710678118654752f)));
    } else if constexpr (EPI == EPI_SHIFT) {
      float hv = 0.5f * v;
      int c1 = ccol + r + 1;
      if (r >= 1 && c1 < Nseq) C[(long)(r - 1) * ldc + c1] += hv;
      int off = Nseq - 1 - r;
      if (ccol >= off) C[(long)r * ldc + (ccol - off)] += hv;
    }
  }
}

// ---------------------------------------------------------------------------
// transpose + fp32->bf16: dst[c*rows + r] = src[r*cols + c]
// ---------------------------------------------------------------------------
__global__ void transpose_bf16_k(const float* __restrict__ src, __bf16* __restrict__ dst,
                                 int rows, int cols)
{
  int idx = blockIdx.x * 256 + threadIdx.x;
  if (idx >= rows * cols) return;
  int r = idx % rows;
  int c = idx / rows;
  dst[idx] = (__bf16)src[(long)r * cols + c];
}

// vT[b,h,d,j] (rows padded to NPAD_, zero-filled) from qkv bf16 v-block
__global__ void vtrans_k(const __bf16* __restrict__ qkvb, __bf16* __restrict__ vTb)
{
  long idx = (long)blockIdx.x * 256 + threadIdx.x;
  const long total = (long)B_ * H_ * DH_ * NPAD_;
  if (idx >= total) return;
  int  j    = (int)(idx % NPAD_);
  long rest = idx / NPAD_;
  int  d    = (int)(rest % DH_);
  long bh   = rest / DH_;
  int  h    = (int)(bh % H_);
  int  b    = (int)(bh / H_);
  __bf16 v = (__bf16)0.f;
  if (j < N_) v = qkvb[((long)b * N_ + j) * (3 * HD_) + 2 * HD_ + h * DH_ + d];
  vTb[idx] = v;
}

// ---------------------------------------------------------------------------
__global__ void embed_k(const float* __restrict__ eeg, const float* __restrict__ Wp,
                        const float* __restrict__ bp, const float* __restrict__ cls,
                        float* __restrict__ x)
{
  int idx = blockIdx.x * 256 + threadIdx.x;
  if (idx >= B_ * N_ * D_) return;
  int d = idx % D_;
  int n = (idx / D_) % N_;
  int b = idx / (D_ * N_);
  float s;
  if (n == 0) {
    s = cls[d];
  } else {
    int nn = n - 1, c = nn / NW_, w = nn % NW_;
    const float* pat = eeg + ((long)b * C_ + c) * T_ + w * PL_;
    const float* wr  = Wp + d * PL_;
    s = bp[d];
#pragma unroll
    for (int p = 0; p < PL_; ++p) s += pat[p] * wr[p];
  }
  x[idx] = s;
}

__global__ void relpos_k(const float* __restrict__ channel_pos,
                         const float* __restrict__ cls_pos,
                         __bf16* __restrict__ rt, __bf16* __restrict__ rc)
{
  int idx = blockIdx.x * 256 + threadIdx.x;
  if (idx >= B_ * N_ * D_) return;
  int d = idx % D_;
  int n = (idx / D_) % N_;
  int b = idx / (D_ * N_);
  float vt, vc;
  if (n == 0) {
    vt = vc = cls_pos[d];
  } else {
    int nn = n - 1, c = nn / NW_, w = nn % NW_;
    float pt = (float)w;
    float pc = channel_pos[b * C_ + c];
    int i = (d < D_ / 2) ? d : d - D_ / 2;
    float freq = __expf(-((float)(2 * i) / (float)D_) * 9.2103403719761836f);
    if (d < D_ / 2) { vt = __sinf(pt * freq); vc = __sinf(pc * freq); }
    else            { vt = __cosf(pt * freq); vc = __cosf(pc * freq); }
  }
  rt[idx] = (__bf16)vt;
  rc[idx] = (__bf16)vc;
}

__global__ void qbias_k(const float* __restrict__ bt, float* __restrict__ qb)
{
  int i = blockIdx.x * 256 + threadIdx.x;
  if (i < 3 * HD_) qb[i] = (i < HD_) ? bt[i] : 0.f;
}

__global__ __launch_bounds__(256)
void layernorm_k(const float* __restrict__ x, __bf16* __restrict__ y,
                 const float* __restrict__ g, const float* __restrict__ b, int rows)
{
  int wid = threadIdx.x >> 5, lane = threadIdx.x & 31;
  int row = blockIdx.x * 8 + wid;
  if (row >= rows) return;
  const float* xr = x + (long)row * D_;
  float v0 = xr[lane], v1 = xr[lane + 32], v2 = xr[lane + 64], v3 = xr[lane + 96];
  float s = v0 + v1 + v2 + v3;
#pragma unroll
  for (int m = 16; m > 0; m >>= 1) s += __shfl_xor(s, m, 32);
  float mean = s * (1.f / (float)D_);
  float d0 = v0 - mean, d1 = v1 - mean, d2 = v2 - mean, d3 = v3 - mean;
  float q = d0 * d0 + d1 * d1 + d2 * d2 + d3 * d3;
#pragma unroll
  for (int m = 16; m > 0; m >>= 1) q += __shfl_xor(q, m, 32);
  float r = rsqrtf(q * (1.f / (float)D_) + 1e-5f);
  __bf16* yr = y + (long)row * D_;
  yr[lane]      = (__bf16)(d0 * r * g[lane]      + b[lane]);
  yr[lane + 32] = (__bf16)(d1 * r * g[lane + 32] + b[lane + 32]);
  yr[lane + 64] = (__bf16)(d2 * r * g[lane + 64] + b[lane + 64]);
  yr[lane + 96] = (__bf16)(d3 * r * g[lane + 96] + b[lane + 96]);
}

__global__ __launch_bounds__(256)
void softmax_k(float* __restrict__ logits, int N)
{
  __shared__ float red[256];
  long row = blockIdx.x;
  float* p = logits + row * (long)N;
  int t = threadIdx.x;
  float mx = -3.4e38f;
  for (int j = t; j < N; j += 256) mx = fmaxf(mx, p[j]);
  red[t] = mx; __syncthreads();
  for (int s = 128; s > 0; s >>= 1) { if (t < s) red[t] = fmaxf(red[t], red[t + s]); __syncthreads(); }
  mx = red[0]; __syncthreads();
  float sum = 0.f;
  for (int j = t; j < N; j += 256) { float e = __expf(p[j] - mx); p[j] = e; sum += e; }
  red[t] = sum; __syncthreads();
  for (int s = 128; s > 0; s >>= 1) { if (t < s) red[t] += red[t + s]; __syncthreads(); }
  float inv = 1.f / red[0];
  for (int j = t; j < N; j += 256) p[j] *= inv;
}

__global__ __launch_bounds__(128)
void head_k(const float* __restrict__ x, const float* __restrict__ g,
            const float* __restrict__ bb, const float* __restrict__ W,
            const float* __restrict__ hbias, float* __restrict__ out)
{
  int b = blockIdx.x, t = threadIdx.x;
  __shared__ float sh[128], red[128];
  float v = x[(long)b * N_ * D_ + t];
  red[t] = v; __syncthreads();
  for (int s = 64; s > 0; s >>= 1) { if (t < s) red[t] += red[t + s]; __syncthreads(); }
  float mean = red[0] * (1.f / (float)D_); __syncthreads();
  float d = v - mean;
  red[t] = d * d; __syncthreads();
  for (int s = 64; s > 0; s >>= 1) { if (t < s) red[t] += red[t + s]; __syncthreads(); }
  float r = rsqrtf(red[0] * (1.f / (float)D_) + 1e-5f);
  sh[t] = d * r * g[t] + bb[t];
  __syncthreads();
  if (t < NCLS_) {
    float s = hbias[t];
    for (int k = 0; k < D_; ++k) s += sh[k] * W[k * NCLS_ + t];
    out[b * NCLS_ + t] = s;
  }
}

// ---------------------------------------------------------------------------
static inline dim3 gemm_grid(int M, int Ncols, int batch) {
  return dim3((unsigned)((Ncols + 63) / 64), (unsigned)((M + 31) / 32), (unsigned)batch);
}

extern "C" void kernel_launch(void* const* d_in, const int* in_sizes, int n_in,
                              void* d_out, int out_size, void* d_ws, size_t ws_size,
                              hipStream_t stream)
{
  (void)in_sizes; (void)n_in; (void)out_size; (void)ws_size;

  const float* x_eeg       = (const float*)d_in[0];
  /* d_in[1] = start_times: unused by reference */
  const float* channel_pos = (const float*)d_in[2];
  const float* Wp          = (const float*)d_in[3];
  const float* bp          = (const float*)d_in[4];
  const float* cls_token   = (const float*)d_in[5];
  const float* cls_pos     = (const float*)d_in[6];
  const float* bias_time_e = (const float*)d_in[7];
  const float* ln1_g       = (const float*)d_in[8];
  const float* ln1_b       = (const float*)d_in[9];
  const float* Wqkv        = (const float*)d_in[10];
  const float* Wkrt        = (const float*)d_in[11];
  const float* Wkrc        = (const float*)d_in[12];
  const float* Wo          = (const float*)d_in[13];
  const float* bo          = (const float*)d_in[14];
  const float* ln2_g       = (const float*)d_in[15];
  const float* ln2_b       = (const float*)d_in[16];
  const float* W1          = (const float*)d_in[17];
  const float* b1          = (const float*)d_in[18];
  const float* W2          = (const float*)d_in[19];
  const float* b2          = (const float*)d_in[20];
  const float* head_g      = (const float*)d_in[21];
  const float* head_b      = (const float*)d_in[22];
  const float* head_W      = (const float*)d_in[23];
  const float* head_bias   = (const float*)d_in[24];

  // ---- workspace: fp32 region first (sizes multiple of 4 -> 16B alignment kept)
  float* wsf = (float*)d_ws;
  long of = 0;
  float* x  = wsf + of; of += (long)B_ * N_ * D_;
  float* qb = wsf + of; of += 3 * HD_;
  // ---- bf16 region
  __bf16* wsb = (__bf16*)(wsf + of);
  long ob = 0;
  __bf16* xnb   = wsb + ob; ob += (long)B_ * N_ * D_;
  __bf16* rtb   = wsb + ob; ob += (long)B_ * N_ * D_;
  __bf16* rcb   = wsb + ob; ob += (long)B_ * N_ * D_;
  __bf16* qkvb  = wsb + ob; ob += (long)B_ * N_ * 3 * HD_;
  __bf16* krtb  = wsb + ob; ob += (long)B_ * N_ * HD_;
  __bf16* krcb  = wsb + ob; ob += (long)B_ * N_ * HD_;
  __bf16* aoutb = wsb + ob; ob += (long)B_ * N_ * HD_;
  __bf16* ffhb  = wsb + ob; ob += (long)B_ * N_ * F_;
  __bf16* vTb   = wsb + ob; ob += (long)B_ * H_ * DH_ * NPAD_;
  __bf16* wqkvT = wsb + ob; ob += 3 * HD_ * D_;
  __bf16* wkrtT = wsb + ob; ob += HD_ * D_;
  __bf16* wkrcT = wsb + ob; ob += HD_ * D_;
  __bf16* woT   = wsb + ob; ob += D_ * HD_;
  __bf16* w1T   = wsb + ob; ob += F_ * D_;
  __bf16* w2T   = wsb + ob; ob += D_ * F_;

  float* out  = (float*)d_out;
  float* attn = out + B_ * NCLS_;  // (B,H,N,N) fp32; layer-3 attention survives

  const int tot = B_ * N_ * D_;
  embed_k <<<(tot + 255) / 256, 256, 0, stream>>>(x_eeg, Wp, bp, cls_token, x);
  relpos_k<<<(tot + 255) / 256, 256, 0, stream>>>(channel_pos, cls_pos, rtb, rcb);
  qbias_k <<<6, 256, 0, stream>>>(bias_time_e, qb);

  const int  MR  = B_ * N_;
  const long sAb = (long)N_ * 3 * HD_;
  const long sKb = (long)N_ * HD_;
  const long sVb = (long)H_ * DH_ * NPAD_;
  const long sVh = (long)DH_ * NPAD_;
  const long sCb = (long)H_ * N_ * N_;
  const long sCh = (long)N_ * N_;

  const dim3 sgrid((N_ + 127) / 128, (N_ + 63) / 64, B_ * H_);

  for (int l = 0; l < L_; ++l) {
    transpose_bf16_k<<<(D_ * 3 * HD_ + 255) / 256, 256, 0, stream>>>(
        Wqkv + (long)l * D_ * 3 * HD_, wqkvT, D_, 3 * HD_);
    transpose_bf16_k<<<(D_ * HD_ + 255) / 256, 256, 0, stream>>>(
        Wkrt + (long)l * D_ * HD_, wkrtT, D_, HD_);
    transpose_bf16_k<<<(D_ * HD_ + 255) / 256, 256, 0, stream>>>(
        Wkrc + (long)l * D_ * HD_, wkrcT, D_, HD_);
    transpose_bf16_k<<<(HD_ * D_ + 255) / 256, 256, 0, stream>>>(
        Wo + (long)l * HD_ * D_, woT, HD_, D_);
    transpose_bf16_k<<<(D_ * F_ + 255) / 256, 256, 0, stream>>>(
        W1 + (long)l * D_ * F_, w1T, D_, F_);
    transpose_bf16_k<<<(F_ * D_ + 255) / 256, 256, 0, stream>>>(
        W2 + (long)l * F_ * D_, w2T, F_, D_);

    layernorm_k<<<(MR + 7) / 8, 256, 0, stream>>>(x, xnb, ln1_g + l * D_, ln1_b + l * D_, MR);

    wmma_gemm<__bf16, __bf16, EPI_BIAS><<<gemm_grid(MR, 3 * HD_, 1), 256, 0, stream>>>(
        xnb, D_, 0, 0, wqkvT, D_, 0, 0, qkvb, 3 * HD_, 0, 0,
        MR, 3 * HD_, D_, 1, qb, 0);

    vtrans_k<<<(int)(((long)B_ * H_ * DH_ * NPAD_ + 255) / 256), 256, 0, stream>>>(qkvb, vTb);

    wmma_gemm<__bf16, __bf16, EPI_STORE><<<gemm_grid(MR, HD_, 1), 256, 0, stream>>>(
        rtb, D_, 0, 0, wkrtT, D_, 0, 0, krtb, HD_, 0, 0, MR, HD_, D_, 1, nullptr, 0);
    wmma_gemm<__bf16, __bf16, EPI_STORE><<<gemm_grid(MR, HD_, 1), 256, 0, stream>>>(
        rcb, D_, 0, 0, wkrcT, D_, 0, 0, krcb, HD_, 0, 0, MR, HD_, D_, 1, nullptr, 0);

    // AC = q @ k^T (LDS-staged, async); then fused rel-shift BD scatters
    score_gemm<EPI_STORE><<<sgrid, 256, 0, stream>>>(
        qkvb, 3 * HD_, sAb, DH_, qkvb + HD_, 3 * HD_, sAb, DH_, attn, N_);
    score_gemm<EPI_SHIFT><<<sgrid, 256, 0, stream>>>(
        qkvb, 3 * HD_, sAb, DH_, krtb, HD_, sKb, DH_, attn, N_);
    score_gemm<EPI_SHIFT><<<sgrid, 256, 0, stream>>>(
        qkvb, 3 * HD_, sAb, DH_, krcb, HD_, sKb, DH_, attn, N_);

    softmax_k<<<B_ * H_ * N_, 256, 0, stream>>>(attn, N_);

    wmma_gemm<float, __bf16, EPI_STORE><<<gemm_grid(N_, DH_, B_ * H_), 256, 0, stream>>>(
        attn, N_, sCb, sCh, vTb, NPAD_, sVb, sVh,
        aoutb, HD_, sKb, DH_, N_, DH_, N_, H_, nullptr, 0);

    wmma_gemm<__bf16, float, EPI_BIAS_RES><<<gemm_grid(MR, D_, 1), 256, 0, stream>>>(
        aoutb, HD_, 0, 0, woT, HD_, 0, 0, x, D_, 0, 0,
        MR, D_, HD_, 1, bo + l * D_, 0);

    layernorm_k<<<(MR + 7) / 8, 256, 0, stream>>>(x, xnb, ln2_g + l * D_, ln2_b + l * D_, MR);

    wmma_gemm<__bf16, __bf16, EPI_GELU_BIAS><<<gemm_grid(MR, F_, 1), 256, 0, stream>>>(
        xnb, D_, 0, 0, w1T, D_, 0, 0, ffhb, F_, 0, 0,
        MR, F_, D_, 1, b1 + l * F_, 0);

    wmma_gemm<__bf16, float, EPI_BIAS_RES><<<gemm_grid(MR, D_, 1), 256, 0, stream>>>(
        ffhb, F_, 0, 0, w2T, F_, 0, 0, x, D_, 0, 0,
        MR, D_, F_, 1, b2 + l * D_, 0);
  }

  head_k<<<B_, 128, 0, stream>>>(x, head_g, head_b, head_W, head_bias, out);
}
